// ScorePredictor_28286654611936
// MI455X (gfx1250) — compile-verified, matrix operand order checked by reading, not run on hardware
//
#include <hip/hip_runtime.h>
#include <hip/hip_bf16.h>
#include <stdint.h>

// ---------------------------------------------------------------------------
// Edge-MLP score predictor for MI455X (gfx1250, wave32, WMMA)
//   data  = [x[src] | x[dst] | e]          : [E, 384]  (fp32 -> bf16)
//   h     = relu(data @ W1^T + b1)         : [E, 256]  (bf16 WMMA, f32 accum)
//   score = h @ W2^T + b2                  : [E, 1]    (fused VALU epilogue)
//
// v2: TM=128 edges/block; each wave computes 32 edges x 128 hidden so every
// B-fragment register load feeds TWO WMMAs (halves L2->WGP B traffic/FLOP).
// ---------------------------------------------------------------------------

typedef __attribute__((ext_vector_type(16))) __bf16 v16bf;
typedef __attribute__((ext_vector_type(8)))  float  v8f;

#define F_DIM   128
#define K_DIM   384          // 3*F
#define HID     256
#define TM      128          // edges per block
#define LDA     392          // A-tile row stride in bf16 (384 + 8 pad, 16B-aligned)
#define KSTEPS  12           // 384 / 32
#define NTILES  16           // 256 / 16

union FragBF {
    uint4 u4[2];
    v16bf v;
};

__device__ __forceinline__ unsigned short f2bf(float f) {
    union { float f; unsigned u; } x; x.f = f;
    unsigned r = x.u + 0x7FFFu + ((x.u >> 16) & 1u);   // round-to-nearest-even
    return (unsigned short)(r >> 16);
}

// ---------------------------------------------------------------------------
// Kernel 1: pack W1 [256][384] f32 -> bf16 in per-lane WMMA-B fragment order.
// Fragment (ks, nt): 32 lanes x 16 bf16, lane-contiguous so each lane does two
// b128 loads. Assumed B (32x16 KxN) layout mirrors the documented 16-bit A
// layout: lane l -> N = nt*16 + (l&15); lane-half selects K offset +8/+24;
// VGPR v<4 -> K = ks*32 + half*8 + 2v+j ; v>=4 -> K = ks*32 + 16 + half*8 + ...
// ---------------------------------------------------------------------------
__global__ void pack_w1_kernel(const float* __restrict__ W1,
                               unsigned short* __restrict__ Bpk) {
    int idx = blockIdx.x * blockDim.x + threadIdx.x;
    if (idx >= KSTEPS * NTILES * 32 * 16) return;
    int j    = idx & 1;
    int v    = (idx >> 1) & 7;
    int l    = (idx >> 4) & 31;
    int nt   = (idx >> 9) & 15;
    int ks   = idx >> 13;
    int n    = nt * 16 + (l & 15);
    int half = l >> 4;
    int k;
    if (v < 4) k = ks * 32 + half * 8 + 2 * v + j;
    else       k = ks * 32 + 16 + half * 8 + 2 * (v - 4) + j;
    Bpk[idx] = f2bf(W1[n * K_DIM + k]);
}

// ---------------------------------------------------------------------------
// Kernel 2: gather -> LDS (bf16) -> WMMA GEMM -> fused relu/W2 epilogue
// Block: 256 threads = 8 waves, 128 edges.
// Wave w: M rows [ (w>>1)*32 , +32 ), N-half (w&1)*128.
// Per wave: 12 K-steps x 8 N-tiles x 2 M-subtiles = 192 WMMAs,
// each B fragment reused for 2 WMMAs.
// ---------------------------------------------------------------------------
__global__ void __launch_bounds__(256)
edge_mlp_kernel(const float* __restrict__ x,
                const float* __restrict__ e,
                const long long* __restrict__ src,
                const long long* __restrict__ dst,
                const float* __restrict__ b1,
                const float* __restrict__ W2,
                const float* __restrict__ b2,
                const unsigned short* __restrict__ Bpk,
                float* __restrict__ out,
                long long E) {
    __shared__ unsigned short sA[TM * LDA];   // ~100 KB bf16 A-tile
    __shared__ float sPart[2][TM];

    const long long ebase = (long long)blockIdx.x * TM;

    // ---- Gather + fp32->bf16 convert into LDS A-tile --------------------
    for (int t = threadIdx.x; t < TM * (K_DIM / 4); t += 256) {
        int row = t / (K_DIM / 4);
        int c4  = (t % (K_DIM / 4)) * 4;
        long long eidx = ebase + row;
        float4 val = make_float4(0.f, 0.f, 0.f, 0.f);
        if (eidx < E) {
            if (c4 < F_DIM) {
                val = *(const float4*)(x + (size_t)src[eidx] * F_DIM + c4);
            } else if (c4 < 2 * F_DIM) {
                val = *(const float4*)(x + (size_t)dst[eidx] * F_DIM + (c4 - F_DIM));
            } else {
                val = *(const float4*)(e + (size_t)eidx * F_DIM + (c4 - 2 * F_DIM));
            }
        }
        uint2 pk;
        pk.x = (unsigned)f2bf(val.x) | ((unsigned)f2bf(val.y) << 16);
        pk.y = (unsigned)f2bf(val.z) | ((unsigned)f2bf(val.w) << 16);
        *(uint2*)&sA[row * LDA + c4] = pk;
    }
    __syncthreads();

    // ---- WMMA GEMM: [32 x 384] @ [384 x 128] per wave --------------------
    const int lane   = threadIdx.x & 31;
    const int wave   = threadIdx.x >> 5;
    const int m_base = (wave >> 1) * 32;  // 0,32,64,96
    const int n_half = wave & 1;          // 0..1
    const int half   = lane >> 4;         // lane-half for K striping
    const int row0   = m_base + (lane & 15);
    const int row1   = row0 + 16;

    v8f acc0[8], acc1[8];
#pragma unroll
    for (int i = 0; i < 8; ++i) { acc0[i] = (v8f)(0.0f); acc1[i] = (v8f)(0.0f); }

#pragma unroll
    for (int ks = 0; ks < KSTEPS; ++ks) {
        // A fragments for the two 16-edge M-subtiles of this wave.
        // elems 0..7  <- K = ks*32 + half*8 .. +7   (VGPR0-3)
        // elems 8..15 <- K = ks*32 + 16 + half*8 .. (VGPR4-7)
        FragBF a0, a1;
        const uint4* pa0 = (const uint4*)&sA[row0 * LDA + ks * 32 + half * 8];
        const uint4* pa1 = (const uint4*)&sA[row1 * LDA + ks * 32 + half * 8];
        a0.u4[0] = pa0[0];
        a0.u4[1] = pa0[2];   // +16 bf16 = +32 bytes
        a1.u4[0] = pa1[0];
        a1.u4[1] = pa1[2];
#pragma unroll
        for (int nt = 0; nt < 8; ++nt) {
            const int ntg = n_half * 8 + nt;
            const uint4* pb =
                (const uint4*)(Bpk + (((size_t)(ks * 16 + ntg) * 32 + lane) << 4));
            FragBF b;
            b.u4[0] = pb[0];
            b.u4[1] = pb[1];
            acc0[nt] = __builtin_amdgcn_wmma_f32_16x16x32_bf16(
                false, a0.v, false, b.v, (short)0, acc0[nt], false, false);
            acc1[nt] = __builtin_amdgcn_wmma_f32_16x16x32_bf16(
                false, a1.v, false, b.v, (short)0, acc1[nt], false, false);
        }
    }

    // ---- Fused epilogue: relu(acc + b1) * W2, reduce over N --------------
    // C layout: VGPR r, lane l -> M = r + 8*(l>=16), N = l&15 (per N-tile).
    float p0[8], p1[8];
#pragma unroll
    for (int r = 0; r < 8; ++r) { p0[r] = 0.0f; p1[r] = 0.0f; }
#pragma unroll
    for (int nt = 0; nt < 8; ++nt) {
        const int n  = n_half * 128 + nt * 16 + (lane & 15);
        const float bb = b1[n];
        const float w2 = W2[n];
#pragma unroll
        for (int r = 0; r < 8; ++r) {
            float v0 = acc0[nt][r] + bb;
            float v1 = acc1[nt][r] + bb;
            v0 = v0 > 0.0f ? v0 : 0.0f;
            v1 = v1 > 0.0f ? v1 : 0.0f;
            p0[r] += v0 * w2;
            p1[r] += v1 * w2;
        }
    }
    // 16-lane reduction within each lane-half (wave32 shuffles)
#pragma unroll
    for (int m = 1; m < 16; m <<= 1) {
#pragma unroll
        for (int r = 0; r < 8; ++r) {
            p0[r] += __shfl_xor(p0[r], m, 32);
            p1[r] += __shfl_xor(p1[r], m, 32);
        }
    }
    if ((lane & 15) == 0) {
#pragma unroll
        for (int r = 0; r < 8; ++r) {
            sPart[n_half][m_base + half * 8 + r]      = p0[r];
            sPart[n_half][m_base + 16 + half * 8 + r] = p1[r];
        }
    }
    __syncthreads();

    // ---- Combine the two N-halves, add b2, store -------------------------
    if (threadIdx.x < TM) {
        long long eidx = ebase + threadIdx.x;
        if (eidx < E)
            out[eidx] = sPart[0][threadIdx.x] + sPart[1][threadIdx.x] + b2[0];
    }
}

// ---------------------------------------------------------------------------
extern "C" void kernel_launch(void* const* d_in, const int* in_sizes, int n_in,
                              void* d_out, int out_size, void* d_ws, size_t ws_size,
                              hipStream_t stream) {
    const float*     x   = (const float*)d_in[0];
    const float*     e   = (const float*)d_in[1];
    const long long* src = (const long long*)d_in[2];   // int64 indices
    const long long* dst = (const long long*)d_in[3];
    const float*     W1  = (const float*)d_in[4];
    const float*     b1  = (const float*)d_in[5];
    const float*     W2  = (const float*)d_in[6];
    const float*     b2  = (const float*)d_in[7];
    float*           out = (float*)d_out;

    unsigned short* Bpk = (unsigned short*)d_ws;        // 192 KB packed W1 (bf16)
    const long long E   = in_sizes[2];

    const int pack_elems = KSTEPS * NTILES * 32 * 16;   // 98304
    pack_w1_kernel<<<(pack_elems + 255) / 256, 256, 0, stream>>>(W1, Bpk);

    const int blocks = (int)((E + TM - 1) / TM);
    edge_mlp_kernel<<<blocks, 256, 0, stream>>>(x, e, src, dst, b1, W2, b2,
                                                Bpk, out, E);
}
